// CIN_4612794876007
// MI455X (gfx1250) — matrix operand order repaired and checked by reading, not compiled, not run on hardware
//
#include <hip/hip_runtime.h>
#include <hip/hip_bf16.h>

// ---------------------------------------------------------------------------
// CIN (xDeepFM) on gfx1250: z-on-the-fly bf16 WMMA GEMM.
//   Layer 0: [B*D=32768, K=1521] x [1521,128]
//   Layer 1: [B*D=32768, K=4992] x [4992,128]
// A (z) generated in LDS in WMMA fragment layout; W pre-converted to bf16
// fragment layout in workspace (L2-resident, reused by every block).
// Layer 1 fast path: X1 stored transposed [d][n] in LDS so each group of 8
// K-elements is one ds_load_b128 + one hoisted x0 load, no divisions.
// ---------------------------------------------------------------------------

typedef __attribute__((ext_vector_type(16))) __bf16 v16bf;
typedef __attribute__((ext_vector_type(8)))  float  v8f;
typedef __attribute__((ext_vector_type(4)))  unsigned int v4u;

#define GB    4     // batch elements per workgroup (M rows = GB*16)
#define CB    4     // K-chunks (of 32) staged per round
#define NTILE 8     // 8 N-tiles of 16 -> 128 outputs, one per wave
#define BLOCK 256

__device__ __forceinline__ unsigned short f2bf(float f) {   // fp32 -> bf16 RNE
  unsigned u = __builtin_bit_cast(unsigned, f);
  u += 0x7FFFu + ((u >> 16) & 1u);
  return (unsigned short)(u >> 16);
}

// ---------------------------------------------------------------------------
// W fp32 [KACT,128] -> bf16 fragments [NCH chunks][8 tiles][32 lanes][16 elems]
// B-fragment layout (32x16, K x N): lane = half*16 + N, elem e -> K = 16*(e>=8) + 8*half + (e&7)
// ---------------------------------------------------------------------------
template<int KACT, int NCH>
__global__ __launch_bounds__(256) void convert_w(const float* __restrict__ W,
                                                 unsigned short* __restrict__ Wf) {
  int idx = blockIdx.x * 256 + threadIdx.x;       // one bf16 element
  if (idx >= NCH * NTILE * 32 * 16) return;
  int e    = idx & 15;
  int L    = (idx >> 4) & 31;
  int t    = (idx >> 9) & 7;
  int c    = idx >> 12;                            // 8*32*16 = 4096 per chunk
  int k    = c * 32 + (L >> 4) * 8 + ((e & 8) << 1) + (e & 7);
  int col  = t * 16 + (L & 15);
  unsigned short v = 0;
  if (k < KACT) v = f2bf(W[(size_t)k * 128 + col]);
  Wf[idx] = v;
}

// ---------------------------------------------------------------------------
// One CIN layer for GB batch elements.
//   sx0 : [GB][39][16] fp32        (X0, n-major)
//   sx1 : [GB][16][128] bf16       (X1 TRANSPOSED, d-major; layer0 -> layer1)
//   zbuf: double-buffered [2][CB][GB][32 lanes][16 elems] bf16 A-fragments
// A-fragment layout (16x32, M x K): lane = half*16 + d,
//   elem e -> K = 16*(e>=8) + 8*half + (e&7)   (ISA 16-bit A 16x32 table)
// ---------------------------------------------------------------------------
template<int MI, int NCH, int KACT, bool LAYER1>
__device__ __forceinline__ void cin_layer(const float* __restrict__ sx0,
                                          unsigned short* __restrict__ sx1,
                                          unsigned short* __restrict__ zbuf,
                                          const unsigned short* __restrict__ wfrag,
                                          float* __restrict__ out,
                                          int b0, int tid, int outoff) {
  constexpr int NR = NCH / CB;
  const int wave = tid >> 5;
  const int lane = tid & 31;

  v8f acc[GB];
#pragma unroll
  for (int g = 0; g < GB; ++g) acc[g] = v8f{};

  // cooperative z-fragment generation: 512 fragment-lanes per round / 256 thr
  auto gen = [&](int round, int buf) {
    unsigned short* zb = zbuf + buf * (CB * GB * 512);
    const int kbase = round * (CB * 32);
#pragma unroll
    for (int i = 0; i < 2; ++i) {
      int fl = tid + i * BLOCK;                 // 0..511
      int cc = fl >> 7;                         // chunk-in-round
      int g  = (fl >> 5) & (GB - 1);
      int L  = fl & 31;
      int half = L >> 4, d = L & 15;
      int kc = kbase + cc * 32 + half * 8;
      unsigned pk[8];
      if constexpr (LAYER1) {
        // MI=128: per 8-aligned group, m constant, n consecutive -> vector load
#pragma unroll
        for (int go = 0; go < 2; ++go) {
          int pbase = kc + go * 16;             // 8-aligned
          int m  = pbase >> 7;
          int n0 = pbase & 127;                 // multiple of 8
          float a = sx0[g * 624 + m * 16 + d];
          v4u xi = *(const v4u*)(sx1 + g * 2048 + d * 128 + n0); // 8 bf16
#pragma unroll
          for (int j = 0; j < 4; ++j) {
            float xlo = __builtin_bit_cast(float, xi[j] << 16);
            float xhi = __builtin_bit_cast(float, xi[j] & 0xFFFF0000u);
            unsigned lo = f2bf(a * xlo);
            unsigned hi = f2bf(a * xhi);
            pk[go * 4 + j] = lo | (hi << 16);
          }
        }
      } else {
#pragma unroll
        for (int e = 0; e < 16; ++e) {
          int p = kc + ((e & 8) << 1) + (e & 7);  // global K index
          unsigned short zv = 0;
          if (p < KACT) {
            int m = p / MI;                        // constant divisor
            int n = p - m * MI;
            float a  = sx0[g * 624 + m * 16 + d];
            float xi = sx0[g * 624 + n * 16 + d];
            zv = f2bf(a * xi);
          }
          if (e & 1) pk[e >> 1] |= (unsigned)zv << 16;
          else       pk[e >> 1]  = (unsigned)zv;
        }
      }
      v4u* dst = (v4u*)(zb + (size_t)fl * 16);
      v4u lo4 = {pk[0], pk[1], pk[2], pk[3]};
      v4u hi4 = {pk[4], pk[5], pk[6], pk[7]};
      dst[0] = lo4;
      dst[1] = hi4;
    }
  };

  gen(0, 0);
  __syncthreads();

  for (int r = 0; r < NR; ++r) {
    const int buf = r & 1;
    const v16bf* zfrag = (const v16bf*)zbuf + (size_t)buf * (CB * GB * 32);
    const v16bf* wf    = (const v16bf*)wfrag;
#pragma unroll
    for (int cc = 0; cc < CB; ++cc) {
      const int chunk = r * CB + cc;
      v16bf bfrag = wf[(size_t)(chunk * NTILE + wave) * 32 + lane];
#pragma unroll
      for (int g = 0; g < GB; ++g) {
        v16bf afrag = zfrag[(size_t)(cc * GB + g) * 32 + lane];
        acc[g] = __builtin_amdgcn_wmma_f32_16x16x32_bf16(
            false, afrag, false, bfrag, (short)0, acc[g], false, false);
      }
    }
    if (r + 1 < NR) gen(r + 1, buf ^ 1);
    __syncthreads();
  }

  // Epilogue: C layout lane = half*16 + n, vgpr v -> M(d) = v + 8*half
  const int half = lane >> 4;
  const int o    = wave * 16 + (lane & 15);
#pragma unroll
  for (int g = 0; g < GB; ++g) {
    if (!LAYER1) {                  // X1 -> LDS bf16, TRANSPOSED [d][n]
#pragma unroll
      for (int v = 0; v < 8; ++v)
        sx1[g * 2048 + (v + 8 * half) * 128 + o] = f2bf(acc[g][v]);
    }
    float s = 0.f;
#pragma unroll
    for (int v = 0; v < 8; ++v) s += acc[g][v]; // partial sum over 8 d's
    s += __shfl_xor(s, 16, 32);                 // combine halves (wave32)
    if (lane < 16)
      out[(size_t)(b0 + g) * 256 + outoff + wave * 16 + lane] = s;
  }
}

__global__ __launch_bounds__(256) void cin_main(const float* __restrict__ X,
                                                const unsigned short* __restrict__ W0f,
                                                const unsigned short* __restrict__ W1f,
                                                float* __restrict__ out) {
  __shared__ __align__(32) float          sx0[GB * 39 * 16];
  __shared__ __align__(32) unsigned short sx1[GB * 16 * 128];
  __shared__ __align__(32) unsigned short zbuf[2 * CB * GB * 512];

  const int tid = threadIdx.x;
  const int b0  = blockIdx.x * GB;

  for (int idx = tid; idx < GB * 39 * 16; idx += BLOCK)
    sx0[idx] = X[(size_t)b0 * 624 + idx];
  __syncthreads();

  cin_layer< 39,  48, 1521, false>(sx0, sx1, zbuf, W0f, out, b0, tid,   0);
  __syncthreads();
  cin_layer<128, 156, 4992, true >(sx0, sx1, zbuf, W1f, out, b0, tid, 128);
}

extern "C" void kernel_launch(void* const* d_in, const int* in_sizes, int n_in,
                              void* d_out, int out_size, void* d_ws, size_t ws_size,
                              hipStream_t stream) {
  const float* X  = (const float*)d_in[0];   // [2048,39,16]
  const float* W0 = (const float*)d_in[1];   // [1521,128]
  const float* W1 = (const float*)d_in[2];   // [4992,128]

  unsigned short* W0f = (unsigned short*)d_ws;             // 48*4096 bf16
  unsigned short* W1f = W0f + 48 * 4096;                   // 156*4096 bf16

  convert_w<1521,  48><<<( 48 * 4096 + 255) / 256, 256, 0, stream>>>(W0, W0f);
  convert_w<4992, 156><<<(156 * 4096 + 255) / 256, 256, 0, stream>>>(W1, W1f);

  cin_main<<<2048 / GB, BLOCK, 0, stream>>>(X, W0f, W1f, (float*)d_out);
}
